// Projector_27650999451950
// MI455X (gfx1250) — 8-bit matrix operand order repaired and checked by reading, compiled, not run on hardware
//
#include <hip/hip_runtime.h>
#include <math.h>


typedef __attribute__((ext_vector_type(8))) int v8i;

#define M_TOK 2048
#define KDIM  4096
#define NDIM  4096
#define KSTAGE 64
#define LDS_STRIDE 80

__device__ inline float wred_sum(float v) {
#pragma unroll
  for (int o = 16; o > 0; o >>= 1) v += __shfl_xor(v, o, 32);
  return v;
}
__device__ inline float wred_max(float v) {
#pragma unroll
  for (int o = 16; o > 0; o >>= 1) v = fmaxf(v, __shfl_xor(v, o, 32));
  return v;
}
__device__ inline float block_sum(float v, float* sm, int tid) {
  v = wred_sum(v);
  if ((tid & 31) == 0) sm[tid >> 5] = v;
  __syncthreads();
  float t = 0.f;
#pragma unroll
  for (int i = 0; i < 8; ++i) t += sm[i];
  __syncthreads();
  return t;
}
__device__ inline float block_max(float v, float* sm, int tid) {
  v = wred_max(v);
  if ((tid & 31) == 0) sm[tid >> 5] = v;
  __syncthreads();
  float t = sm[0];
#pragma unroll
  for (int i = 1; i < 8; ++i) t = fmaxf(t, sm[i]);
  __syncthreads();
  return t;
}

template <bool DO_LN>
__global__ __launch_bounds__(256)
void rowquant_kernel(const float* __restrict__ X,
                     const float* __restrict__ gamma,
                     const float* __restrict__ beta,
                     signed char* __restrict__ Q,
                     float* __restrict__ scale,
                     int K) {
  __shared__ float sm[8];
  const int row = blockIdx.x;
  const int tid = threadIdx.x;
  const float* x = X + (size_t)row * K;

  float v[16];
  const float4* xv = (const float4*)(x + tid * 16);
#pragma unroll
  for (int g = 0; g < 4; ++g) {
    float4 f = xv[g];
    v[4 * g + 0] = f.x; v[4 * g + 1] = f.y;
    v[4 * g + 2] = f.z; v[4 * g + 3] = f.w;
  }

  if (DO_LN) {
    float s = 0.f, ss = 0.f;
#pragma unroll
    for (int j = 0; j < 16; ++j) { s += v[j]; ss += v[j] * v[j]; }
    const float mu  = block_sum(s,  sm, tid) * (1.0f / (float)K);
    const float ex2 = block_sum(ss, sm, tid) * (1.0f / (float)K);
    const float rs  = rsqrtf(fmaxf(ex2 - mu * mu, 0.f) + 1e-5f);
#pragma unroll
    for (int j = 0; j < 16; ++j) {
      const int i = tid * 16 + j;
      v[j] = (v[j] - mu) * rs * gamma[i] + beta[i];
    }
  }

  float am = 0.f;
#pragma unroll
  for (int j = 0; j < 16; ++j) am = fmaxf(am, fabsf(v[j]));
  am = block_max(am, sm, tid);
  const float s   = fmaxf(am * (1.0f / 127.0f), 1e-8f);
  const float inv = 1.0f / s;

  int w[4];
#pragma unroll
  for (int g = 0; g < 4; ++g) {
    int q[4];
#pragma unroll
    for (int j = 0; j < 4; ++j) {
      int qi = (int)rintf(v[4 * g + j] * inv);
      qi = qi > 127 ? 127 : (qi < -127 ? -127 : qi);
      q[j] = qi;
    }
    w[g] = (q[0] & 255) | ((q[1] & 255) << 8) | ((q[2] & 255) << 16) | (q[3] << 24);
  }
  int4 packed = make_int4(w[0], w[1], w[2], w[3]);
  *(int4*)(Q + (size_t)row * K + tid * 16) = packed;
  if (tid == 0) scale[row] = s;
}

__global__ __launch_bounds__(256)
void gemm_i8_kernel(const signed char* __restrict__ Aq,
                    const float* __restrict__ sx,
                    const signed char* __restrict__ Wq,
                    const float* __restrict__ sw,
                    const float* __restrict__ bias,
                    float* __restrict__ out,
                    int M, int N, int K, int do_gelu) {
  __shared__ char As[2][128 * LDS_STRIDE];
  __shared__ char Bs[2][128 * LDS_STRIDE];

  const int tid  = threadIdx.x;
  const int lane = tid & 31;
  const int wave = tid >> 5;
  const int wm = wave >> 2;
  const int wn = wave & 3;
  const int m0 = blockIdx.y * 128;
  const int n0 = blockIdx.x * 128;
  const int hi  = lane >> 4;
  const int l15 = lane & 15;

  const int srow = tid >> 1;
  const int skq  = (tid & 1) * 32;
  const signed char* gA = Aq + (size_t)(m0 + srow) * K + skq;
  const signed char* gB = Wq + (size_t)(n0 + srow) * K + skq;
  const int sdst = srow * LDS_STRIDE + skq;

  int4 ra0 = *(const int4*)(gA);
  int4 ra1 = *(const int4*)(gA + 16);
  int4 rb0 = *(const int4*)(gB);
  int4 rb1 = *(const int4*)(gB + 16);
  *(int4*)&As[0][sdst]      = ra0;
  *(int4*)&As[0][sdst + 16] = ra1;
  *(int4*)&Bs[0][sdst]      = rb0;
  *(int4*)&Bs[0][sdst + 16] = rb1;
  __syncthreads();

  v8i acc[4][2] = {};
  const int nstages = K / KSTAGE;

  for (int s = 0; s < nstages; ++s) {
    const int cur = s & 1;
    if (s + 1 < nstages) {
      const size_t go = (size_t)(s + 1) * KSTAGE;
      ra0 = *(const int4*)(gA + go);
      ra1 = *(const int4*)(gA + go + 16);
      rb0 = *(const int4*)(gB + go);
      rb1 = *(const int4*)(gB + go + 16);
      if (s + 2 < nstages) {
        __builtin_prefetch(gA + go + KSTAGE, 0, 3);
        __builtin_prefetch(gB + go + KSTAGE, 0, 3);
      }
    }

    v8i a[4], b[2];
#pragma unroll
    for (int i = 0; i < 4; ++i) {
      const char* p = &As[cur][(wm * 64 + 16 * i + l15) * LDS_STRIDE + hi * 16];
      const int2 d0 = *(const int2*)(p + 0);
      const int2 d1 = *(const int2*)(p + 8);
      const int2 d2 = *(const int2*)(p + 32);
      const int2 d3 = *(const int2*)(p + 40);
      a[i][0] = d0.x; a[i][1] = d0.y; a[i][2] = d1.x; a[i][3] = d1.y;
      a[i][4] = d2.x; a[i][5] = d2.y; a[i][6] = d3.x; a[i][7] = d3.y;
    }
#pragma unroll
    for (int j = 0; j < 2; ++j) {
      const char* p = &Bs[cur][(wn * 32 + 16 * j + l15) * LDS_STRIDE + hi * 16];
      const int4 lo = *(const int4*)(p + 0);
      const int4 h4 = *(const int4*)(p + 32);
      b[j][0] = lo.x; b[j][1] = lo.y; b[j][2] = lo.z; b[j][3] = lo.w;
      b[j][4] = h4.x; b[j][5] = h4.y; b[j][6] = h4.z; b[j][7] = h4.w;
    }

#pragma unroll
    for (int i = 0; i < 4; ++i)
#pragma unroll
      for (int j = 0; j < 2; ++j)
        acc[i][j] = __builtin_amdgcn_wmma_i32_16x16x64_iu8(
             true, a[i],  true, b[j], acc[i][j],
             false,  false);

    if (s + 1 < nstages) {
      const int nxt = cur ^ 1;
      *(int4*)&As[nxt][sdst]      = ra0;
      *(int4*)&As[nxt][sdst + 16] = ra1;
      *(int4*)&Bs[nxt][sdst]      = rb0;
      *(int4*)&Bs[nxt][sdst + 16] = rb1;
    }
    __syncthreads();
  }

  const int mofs = hi * 8;
#pragma unroll
  for (int i = 0; i < 4; ++i) {
#pragma unroll
    for (int j = 0; j < 2; ++j) {
      const int n = n0 + wn * 32 + 16 * j + l15;
      const float swn = sw[n];
      const float bn  = bias[n];
#pragma unroll
      for (int r = 0; r < 8; ++r) {
        const int m = m0 + wm * 64 + 16 * i + mofs + r;
        float vv = (float)acc[i][j][r] * sx[m] * swn + bn;
        if (do_gelu) vv = 0.5f * vv * (1.0f + erff(vv * 0.70710678118654752f));
        out[(size_t)m * N + n] = vv;
      }
    }
  }
}

extern "C" void kernel_launch(void* const* d_in, const int* in_sizes, int n_in,
                              void* d_out, int out_size, void* d_ws, size_t ws_size,
                              hipStream_t stream) {
  (void)in_sizes; (void)n_in; (void)out_size; (void)ws_size;
  const float* vit   = (const float*)d_in[0];
  const float* gamma = (const float*)d_in[1];
  const float* beta  = (const float*)d_in[2];
  const float* w1    = (const float*)d_in[3];
  const float* b1    = (const float*)d_in[4];
  const float* w3    = (const float*)d_in[5];
  const float* b3    = (const float*)d_in[6];
  float* out = (float*)d_out;

  char* ws = (char*)d_ws;
  auto align256 = [](size_t x) { return (x + 255) & ~(size_t)255; };
  size_t off = 0;
  signed char* A1q = (signed char*)(ws + off); off = align256(off + (size_t)M_TOK * KDIM);
  signed char* W1q = (signed char*)(ws + off); off = align256(off + (size_t)NDIM * KDIM);
  signed char* W3q = (signed char*)(ws + off); off = align256(off + (size_t)NDIM * KDIM);
  signed char* A2q = (signed char*)(ws + off); off = align256(off + (size_t)M_TOK * KDIM);
  float* H   = (float*)(ws + off); off = align256(off + (size_t)M_TOK * NDIM * 4);
  float* sx1 = (float*)(ws + off); off = align256(off + (size_t)M_TOK * 4);
  float* sx2 = (float*)(ws + off); off = align256(off + (size_t)M_TOK * 4);
  float* sw1 = (float*)(ws + off); off = align256(off + (size_t)NDIM * 4);
  float* sw3 = (float*)(ws + off); off = align256(off + (size_t)NDIM * 4);

  rowquant_kernel<true><<<M_TOK, 256, 0, stream>>>(vit, gamma, beta, A1q, sx1, KDIM);
  rowquant_kernel<false><<<NDIM, 256, 0, stream>>>(w1, nullptr, nullptr, W1q, sw1, KDIM);
  rowquant_kernel<false><<<NDIM, 256, 0, stream>>>(w3, nullptr, nullptr, W3q, sw3, KDIM);
  dim3 g1(NDIM / 128, M_TOK / 128);
  gemm_i8_kernel<<<g1, 256, 0, stream>>>(A1q, sx1, W1q, sw1, b1, H,
                                         M_TOK, NDIM, KDIM,  1);
  rowquant_kernel<false><<<M_TOK, 256, 0, stream>>>(H, nullptr, nullptr, A2q, sx2, NDIM);
  gemm_i8_kernel<<<g1, 256, 0, stream>>>(A2q, sx2, W3q, sw3, b3, out,
                                         M_TOK, NDIM, KDIM,  0);
}
